// PatchAttn_3d_68444598829444
// MI455X (gfx1250) — matrix operand order repaired and checked by reading, not compiled, not run on hardware
//
#include <hip/hip_runtime.h>
#include <hip/hip_bf16.h>

typedef __attribute__((ext_vector_type(16))) __bf16 v16bf;
typedef __attribute__((ext_vector_type(8)))  float  v8f;
typedef unsigned short u16;

constexpr int kBatch = 4;
constexpr int kCIn   = 32;
constexpr int kSp    = 64;     // W = H = D
constexpr int kCP    = 2048;   // channels after space_to_channel (32 * 4^3)
constexpr int kNP    = 4096;   // flattened spatial after s2c (16^3)
constexpr int TM = 128, TN = 128, TK = 32;
constexpr int KSTR = 40;       // padded LDS row stride (u16 units): 80B, 16B-aligned

__device__ __forceinline__ u16 f2bf(float f) {
  unsigned u = __builtin_bit_cast(unsigned, f);
  u += 0x7FFFu + ((u >> 16) & 1u);   // round-to-nearest-even
  return (u16)(u >> 16);
}

union FragU  { uint4 u[2]; v16bf v; };
union Pack8  { u16 s[8];  uint4 u; };
union Pack16 { u16 s[16]; uint4 u[2]; };

// ---------------------------------------------------------------------------
// Async global->LDS staging (GLOBAL_LOAD_ASYNC_TO_LDS_B128, ASYNCcnt) with a
// synchronous register-staged fallback if the builtin is unavailable.
// Param 1 is 'int __vector(4) addrspace(1)*' per the compiler diagnostic.
// ---------------------------------------------------------------------------
#if defined(__has_builtin) && __has_builtin(__builtin_amdgcn_global_load_async_to_lds_b128)
#define USE_ASYNC_LDS 1
#else
#define USE_ASYNC_LDS 0
#endif

typedef int v4i_vs __attribute__((vector_size(16)));
typedef __attribute__((address_space(1))) v4i_vs* g_v4i_ptr;
typedef __attribute__((address_space(3))) v4i_vs* l_v4i_ptr;

__device__ __forceinline__ void async_copy16B(u16* dst_lds, const u16* src_gl) {
#if USE_ASYNC_LDS
  __builtin_amdgcn_global_load_async_to_lds_b128(
      (g_v4i_ptr)src_gl, (l_v4i_ptr)dst_lds, 0, 0);
#else
  *(uint4*)dst_lds = *(const uint4*)src_gl;
#endif
}

__device__ __forceinline__ void wait_async() {
#if USE_ASYNC_LDS
  asm volatile("s_wait_asynccnt 0x0" ::: "memory");
#endif
}

// ---------------------------------------------------------------------------
// Kernel 1: space_to_channel gather + fp32->bf16 pack into q[B][2048][4096].
// Each thread produces 8 consecutive n (one uint4 store).
// ---------------------------------------------------------------------------
__global__ __launch_bounds__(256) void pack_q_kernel(const float* __restrict__ x,
                                                     u16* __restrict__ q) {
  int gid = blockIdx.x * 256 + threadIdx.x;     // 2^22 threads total
  int d8 = gid & 1;
  int hp = (gid >> 1) & 15;
  int wp = (gid >> 5) & 15;
  int ch = (gid >> 9) & 2047;
  int b  = gid >> 20;
  int c = ch & 31, g4 = (ch >> 5) & 3, g16 = ch >> 7;
  int wi  = 4 * wp + (hp >> 2);
  int hi  = (hp & 3) * 16 + g16;
  int di0 = 32 * d8 + g4;                        // d16 = 8*d8 + j -> di = 4*d16+g4
  const float* xp = x + ((((size_t)(b * kCIn + c) * kSp + wi) * kSp + hi) * kSp + di0);
  Pack8 p;
#pragma unroll
  for (int j = 0; j < 8; ++j) p.s[j] = f2bf(xp[4 * j]);
  size_t n0 = (size_t)((wp * 16 + hp) * 16 + d8 * 8);
  *(uint4*)(q + ((size_t)(b * kCP + ch) * kNP + n0)) = p.u;
}

// ---------------------------------------------------------------------------
// Kernel 2: energy[b] = q[b] (2048x4096) * q[b]^T  -> fp32 (2048x2048).
// 128x128 tile / block, 8 waves (2x4) each own a 64x32 region (4x2 WMMA tiles).
// ---------------------------------------------------------------------------
__global__ __launch_bounds__(256) void gemm1_gram_kernel(const u16* __restrict__ q,
                                                         float* __restrict__ energy) {
  __shared__ __align__(16) u16 As[2][TM * KSTR];
  __shared__ __align__(16) u16 Bs[2][TN * KSTR];
  const int b = blockIdx.z;
  const int rowBase = blockIdx.y * TM;
  const int colBase = blockIdx.x * TN;
  const u16* qb = q + (size_t)b * kCP * kNP;
  const int tid = threadIdx.x;
  const int lane = tid & 31;
  const int wv = tid >> 5;
  const int wr = wv >> 2, wc = wv & 3;

  v8f acc[4][2];
#pragma unroll
  for (int m = 0; m < 4; ++m)
#pragma unroll
    for (int n = 0; n < 2; ++n)
#pragma unroll
      for (int j = 0; j < 8; ++j) acc[m][n][j] = 0.0f;

  auto loadTile = [&](u16* s, int rBase, int k0) {
    int chunk = tid & 3;                         // 4 x 16B chunks per 64B row
    int r0 = tid >> 2;
#pragma unroll
    for (int i = 0; i < 2; ++i) {
      int r = r0 + i * 64;
      async_copy16B(s + r * KSTR + chunk * 8,
                    qb + (size_t)(rBase + r) * kNP + k0 + chunk * 8);
    }
  };

  loadTile(As[0], rowBase, 0);
  loadTile(Bs[0], colBase, 0);
  wait_async();
  __syncthreads();

  const int arow   = lane & 15;
  const int kbaseA = (lane < 16) ? 0 : 8;        // 16-bit A 16x32 layout
  const int koffB  = (lane < 16) ? 0 : 8;        // bf16 B 32x16 layout

  for (int k0 = 0; k0 < kNP; k0 += TK) {
    int cur = (k0 / TK) & 1;
    if (k0 + TK < kNP) {
      loadTile(As[cur ^ 1], rowBase, k0 + TK);
      loadTile(Bs[cur ^ 1], colBase, k0 + TK);
    }
    const u16* A  = As[cur];
    const u16* Bt = Bs[cur];
    v16bf afr[4], bfr[2];
#pragma unroll
    for (int m = 0; m < 4; ++m) {
      const u16* p = A + (wr * 64 + m * 16 + arow) * KSTR;
      FragU f;
      f.u[0] = *(const uint4*)(p + kbaseA);      // K kbase..kbase+7
      f.u[1] = *(const uint4*)(p + kbaseA + 16); // K kbase+16..kbase+23
      afr[m] = f.v;
    }
#pragma unroll
    for (int n = 0; n < 2; ++n) {
      const u16* p = Bt + (wc * 32 + n * 16 + arow) * KSTR + koffB;
      FragU f;
      f.u[0] = *(const uint4*)(p);
      f.u[1] = *(const uint4*)(p + 16);
      bfr[n] = f.v;
    }
#pragma unroll
    for (int m = 0; m < 4; ++m)
#pragma unroll
      for (int n = 0; n < 2; ++n)
        acc[m][n] = __builtin_amdgcn_wmma_f32_16x16x32_bf16(
            false, afr[m], false, bfr[n], (short)0, acc[m][n], false, false);
    wait_async();
    __syncthreads();
  }

  float* eb = energy + (size_t)b * kCP * kCP;
#pragma unroll
  for (int m = 0; m < 4; ++m)
#pragma unroll
    for (int n = 0; n < 2; ++n) {
      int r0 = rowBase + wr * 64 + m * 16 + (lane >> 4) * 8;  // C/D layout
      int cc = colBase + wc * 32 + n * 16 + (lane & 15);
#pragma unroll
      for (int j = 0; j < 8; ++j) eb[(size_t)(r0 + j) * kCP + cc] = acc[m][n][j];
    }
}

// ---------------------------------------------------------------------------
// Kernel 3: attention row softmax.  softmax(rowmax - e) == exp(rowmin - e)/sum.
// One block per (b, row); 256 threads x 8 elements; bf16 output.
// ---------------------------------------------------------------------------
__global__ __launch_bounds__(256) void softmax_kernel(const float* __restrict__ energy,
                                                      u16* __restrict__ attn) {
  const size_t row = blockIdx.x;
  const float* e = energy + row * kCP;
  u16* a = attn + row * kCP;
  const int t = threadIdx.x;
  float v[8];
  float4 x0 = *(const float4*)(e + t * 8);
  float4 x1 = *(const float4*)(e + t * 8 + 4);
  v[0] = x0.x; v[1] = x0.y; v[2] = x0.z; v[3] = x0.w;
  v[4] = x1.x; v[5] = x1.y; v[6] = x1.z; v[7] = x1.w;
  float mn = v[0];
#pragma unroll
  for (int j = 1; j < 8; ++j) mn = fminf(mn, v[j]);
  __shared__ float red[256];
  red[t] = mn;
  __syncthreads();
  for (int s = 128; s > 0; s >>= 1) {
    if (t < s) red[t] = fminf(red[t], red[t + s]);
    __syncthreads();
  }
  mn = red[0];
  __syncthreads();
  float ex[8];
  float sum = 0.0f;
#pragma unroll
  for (int j = 0; j < 8; ++j) { ex[j] = __expf(mn - v[j]); sum += ex[j]; }
  red[t] = sum;
  __syncthreads();
  for (int s = 128; s > 0; s >>= 1) {
    if (t < s) red[t] += red[t + s];
    __syncthreads();
  }
  float inv = 1.0f / red[0];
  Pack8 p;
#pragma unroll
  for (int j = 0; j < 8; ++j) p.s[j] = f2bf(ex[j] * inv);
  *(uint4*)(a + t * 8) = p.u;
}

// ---------------------------------------------------------------------------
// Kernel 4: out = attn[b] (2048x2048) * q[b] (2048x4096); fused epilogue does
// gamma*out + y (y regathered from x) and scatters through channel_to_space.
// ---------------------------------------------------------------------------
__global__ __launch_bounds__(256) void gemm2_out_kernel(const u16* __restrict__ attn,
                                                        const u16* __restrict__ q,
                                                        const float* __restrict__ x,
                                                        const float* __restrict__ gamma,
                                                        float* __restrict__ out) {
  __shared__ __align__(16) u16 As[2][TM * KSTR];
  __shared__ __align__(16) u16 Bs[2][TN * KSTR];
  const int b = blockIdx.z;
  const int rowBase = blockIdx.y * TM;   // ch tile
  const int nBase   = blockIdx.x * TN;   // spatial tile
  const u16* ab = attn + (size_t)b * kCP * kCP;
  const u16* qb = q + (size_t)b * kCP * kNP;
  const int tid = threadIdx.x;
  const int lane = tid & 31;
  const int wv = tid >> 5;
  const int wr = wv >> 2, wc = wv & 3;

  v8f acc[4][2];
#pragma unroll
  for (int m = 0; m < 4; ++m)
#pragma unroll
    for (int n = 0; n < 2; ++n)
#pragma unroll
      for (int j = 0; j < 8; ++j) acc[m][n][j] = 0.0f;

  auto loadA = [&](u16* s, int k0) {
    int chunk = tid & 3;
    int r0 = tid >> 2;
#pragma unroll
    for (int i = 0; i < 2; ++i) {
      int r = r0 + i * 64;
      async_copy16B(s + r * KSTR + chunk * 8,
                    ab + (size_t)(rowBase + r) * kCP + k0 + chunk * 8);
    }
  };
  auto loadB = [&](u16* s, int k0) {   // stage q[k][n] transposed -> LDS[n][k]
    int nchunk = tid & 7;              // 8 chunks of 16 n
    int kk = tid >> 3;                 // 32 k rows
    Pack16 p;
    const u16* src = qb + (size_t)(k0 + kk) * kNP + nBase + nchunk * 16;
    p.u[0] = *(const uint4*)src;
    p.u[1] = *(const uint4*)(src + 8);
#pragma unroll
    for (int i = 0; i < 16; ++i) s[(nchunk * 16 + i) * KSTR + kk] = p.s[i];
  };

  loadA(As[0], 0);
  loadB(Bs[0], 0);
  wait_async();
  __syncthreads();

  const int arow   = lane & 15;
  const int kbaseA = (lane < 16) ? 0 : 8;
  const int koffB  = (lane < 16) ? 0 : 8;

  for (int k0 = 0; k0 < kCP; k0 += TK) {
    int cur = (k0 / TK) & 1;
    if (k0 + TK < kCP) {
      loadA(As[cur ^ 1], k0 + TK);
      loadB(Bs[cur ^ 1], k0 + TK);
    }
    const u16* A  = As[cur];
    const u16* Bt = Bs[cur];
    v16bf afr[4], bfr[2];
#pragma unroll
    for (int m = 0; m < 4; ++m) {
      const u16* p = A + (wr * 64 + m * 16 + arow) * KSTR;
      FragU f;
      f.u[0] = *(const uint4*)(p + kbaseA);
      f.u[1] = *(const uint4*)(p + kbaseA + 16);
      afr[m] = f.v;
    }
#pragma unroll
    for (int n = 0; n < 2; ++n) {
      const u16* p = Bt + (wc * 32 + n * 16 + arow) * KSTR + koffB;
      FragU f;
      f.u[0] = *(const uint4*)(p);
      f.u[1] = *(const uint4*)(p + 16);
      bfr[n] = f.v;
    }
#pragma unroll
    for (int m = 0; m < 4; ++m)
#pragma unroll
      for (int n = 0; n < 2; ++n)
        acc[m][n] = __builtin_amdgcn_wmma_f32_16x16x32_bf16(
            false, afr[m], false, bfr[n], (short)0, acc[m][n], false, false);
    wait_async();
    __syncthreads();
  }

  const float g = gamma[0];
#pragma unroll
  for (int m = 0; m < 4; ++m) {
    int chB = rowBase + wr * 64 + m * 16 + (lane >> 4) * 8;
#pragma unroll
    for (int n = 0; n < 2; ++n) {
      int nn = nBase + wc * 32 + n * 16 + (lane & 15);
      int wp = nn >> 8, hp = (nn >> 4) & 15, d16 = nn & 15;
#pragma unroll
      for (int j = 0; j < 8; ++j) {
        int ch = chB + j;
        int c = ch & 31, g4 = (ch >> 5) & 3, g16 = ch >> 7;
        // y = x via inverse space_to_channel
        int wi = 4 * wp + (hp >> 2);
        int hi = (hp & 3) * 16 + g16;
        int di = 4 * d16 + g4;
        float y = x[(((size_t)(b * kCIn + c) * kSp + wi) * kSp + hi) * kSp + di];
        float z = g * acc[m][n][j] + y;
        // channel_to_space scatter
        int Wo = wp * 4 + (g16 >> 2);
        int Ho = (g16 & 3) * 16 + hp;
        int Do = d16 * 4 + g4;
        out[(((size_t)(b * kCIn + c) * kSp + Wo) * kSp + Ho) * kSp + Do] = z;
      }
    }
  }
}

// ---------------------------------------------------------------------------
extern "C" void kernel_launch(void* const* d_in, const int* in_sizes, int n_in,
                              void* d_out, int out_size, void* d_ws, size_t ws_size,
                              hipStream_t stream) {
  (void)in_sizes; (void)n_in; (void)out_size; (void)ws_size;
  const float* x     = (const float*)d_in[0];
  const float* gamma = (const float*)d_in[1];
  float* out = (float*)d_out;
  char* ws = (char*)d_ws;
  u16*   q      = (u16*)ws;                               // 64 MiB
  float* energy = (float*)(ws + (((size_t)64) << 20));    // 64 MiB
  u16*   attn   = (u16*)(ws + (((size_t)128) << 20));     // 32 MiB

  pack_q_kernel<<<(kBatch * kCP * 512) / 256, 256, 0, stream>>>(x, q);
  gemm1_gram_kernel<<<dim3(kCP / TN, kCP / TM, kBatch), 256, 0, stream>>>(q, energy);
  softmax_kernel<<<kBatch * kCP, 256, 0, stream>>>(energy, attn);
  gemm2_out_kernel<<<dim3(kNP / TN, kCP / TM, kBatch), 256, 0, stream>>>(attn, q, x, gamma, out);
}
